// TransformerBlock_15496242004499
// MI455X (gfx1250) — compile-verified, matrix operand order checked by reading, not run on hardware
//
#include <hip/hip_runtime.h>
#include <hip/hip_bf16.h>
#include <cstdint>

// ---------------------------------------------------------------------------
// Types for CDNA5 WMMA (wave32)
// ---------------------------------------------------------------------------
typedef __bf16 bf16_t;
typedef bf16_t v16bf __attribute__((ext_vector_type(16)));
typedef float  v8f   __attribute__((ext_vector_type(8)));
typedef unsigned u32x4 __attribute__((ext_vector_type(4)));
typedef int      i32x4 __attribute__((ext_vector_type(4)));

union Frag {             // one 16x32 bf16 A/B fragment: 16 bf16 per lane
    u32x4 u[2];
    v16bf v;
};

static __device__ __forceinline__ v8f wmma_bf16(const v16bf& a, const v16bf& b, const v8f& c) {
    return __builtin_amdgcn_wmma_f32_16x16x32_bf16(false, a, false, b, (short)0, c, false, false);
}

#define DIM_     1024
#define HEADS_   16
#define HD_      64
#define T_       2048
#define B_       2
#define TOK_     4096   // B_*T_
#define HID_     4096

// ---------------------------------------------------------------------------
// gfx1250 async global->LDS (ASYNCcnt-tracked) with safe fallback
// ---------------------------------------------------------------------------
#if __has_builtin(__builtin_amdgcn_global_load_async_to_lds_b128) && \
    __has_builtin(__builtin_amdgcn_s_wait_asynccnt)
#define USE_ASYNC_LDS 1
#else
#define USE_ASYNC_LDS 0
#endif

static __device__ __forceinline__ void tile_b128(const bf16_t* g, bf16_t* l) {
#if USE_ASYNC_LDS
    __builtin_amdgcn_global_load_async_to_lds_b128(
        (__attribute__((address_space(1))) i32x4*)g,
        (__attribute__((address_space(3))) i32x4*)l, 0, 0);
#else
    *(u32x4*)l = *(const u32x4*)g;
#endif
}
static __device__ __forceinline__ void tile_wait_prev() {   // <=4 outstanding
#if USE_ASYNC_LDS
    __builtin_amdgcn_s_wait_asynccnt(4);
#endif
}
static __device__ __forceinline__ void tile_wait_all() {
#if USE_ASYNC_LDS
    __builtin_amdgcn_s_wait_asynccnt(0);
#endif
}

// ---------------------------------------------------------------------------
// Butterfly reductions within 16-lane halves: DPP (pure VALU) when available,
// otherwise __shfl_xor (ds_bpermute).
// ---------------------------------------------------------------------------
#if __has_builtin(__builtin_amdgcn_mov_dpp8) && __has_builtin(__builtin_amdgcn_update_dpp)
#define USE_DPP 1
#else
#define USE_DPP 0
#endif

static constexpr unsigned dpp8_sel(int m) {
    unsigned s = 0;
    for (int i = 0; i < 8; ++i) s |= (unsigned)((i ^ m) & 7) << (3 * i);
    return s;
}

template <int M>
static __device__ __forceinline__ float bfly(float x) {
#if USE_DPP
    int i = __builtin_bit_cast(int, x);
    int j;
    if constexpr (M == 8) {
        // xor8 within a 16-lane row == rotate-by-8: DPP16 row_ror:8 (0x128)
        j = __builtin_amdgcn_update_dpp(i, i, 0x128, 0xf, 0xf, true);
    } else {
        j = __builtin_amdgcn_mov_dpp8(i, dpp8_sel(M));
    }
    return __builtin_bit_cast(float, j);
#else
    return __shfl_xor(x, M);
#endif
}
static __device__ __forceinline__ float rmax16(float x) {
    x = fmaxf(x, bfly<1>(x));
    x = fmaxf(x, bfly<2>(x));
    x = fmaxf(x, bfly<4>(x));
    x = fmaxf(x, bfly<8>(x));
    return x;
}
static __device__ __forceinline__ float rsum16(float x) {
    x += bfly<1>(x);
    x += bfly<2>(x);
    x += bfly<4>(x);
    x += bfly<8>(x);
    return x;
}

// ---------------------------------------------------------------------------
// Weight transpose + fp32->bf16 convert:  out[N][K] = (bf16) in[K][N]
// ---------------------------------------------------------------------------
__global__ __launch_bounds__(256)
void transpose_cvt_kernel(const float* __restrict__ in, bf16_t* __restrict__ out,
                          int K, int N) {
    __shared__ bf16_t tile[32][33];
    const int k0 = blockIdx.y * 32, n0 = blockIdx.x * 32;
    const int tx = threadIdx.x & 31, ty = threadIdx.x >> 5;
#pragma unroll
    for (int i = 0; i < 4; ++i) {
        int kk = ty + i * 8;
        tile[kk][tx] = (bf16_t)in[(size_t)(k0 + kk) * N + n0 + tx];
    }
    __syncthreads();
#pragma unroll
    for (int i = 0; i < 4; ++i) {
        int nn = ty + i * 8;
        out[(size_t)(n0 + nn) * K + k0 + tx] = tile[tx][nn];
    }
}

// ---------------------------------------------------------------------------
// V transpose: v[b*T+t][h*64+d] (bf16) -> vt[(b*16+h)*64+d][t] (bf16)
// ---------------------------------------------------------------------------
__global__ __launch_bounds__(256)
void transpose_v_kernel(const bf16_t* __restrict__ v, bf16_t* __restrict__ vt) {
    __shared__ bf16_t tile[32][33];
    const int bh = blockIdx.z;
    const int b = bh >> 4, h = bh & 15;
    const int t0 = blockIdx.x * 32;
    const int d0 = blockIdx.y * 32;
    const int tx = threadIdx.x & 31, ty = threadIdx.x >> 5;
#pragma unroll
    for (int i = 0; i < 4; ++i) {
        int tt = ty + i * 8;
        tile[tt][tx] = v[(size_t)(b * T_ + t0 + tt) * DIM_ + h * HD_ + d0 + tx];
    }
    __syncthreads();
#pragma unroll
    for (int i = 0; i < 4; ++i) {
        int dd = ty + i * 8;
        vt[((size_t)bh * HD_ + d0 + dd) * T_ + t0 + tx] = tile[tx][dd];
    }
}

// ---------------------------------------------------------------------------
// RMSNorm (row = 1024 floats), bf16 output
// ---------------------------------------------------------------------------
__global__ __launch_bounds__(256)
void rmsnorm_kernel(const float* __restrict__ x, const float* __restrict__ w,
                    bf16_t* __restrict__ out) {
    __shared__ float red[8];
    const size_t row = blockIdx.x;
    const float* xr = x + row * DIM_;
    const int tid = threadIdx.x;
    float4 v = *(const float4*)&xr[tid * 4];
    float s = v.x * v.x + v.y * v.y + v.z * v.z + v.w * v.w;
#pragma unroll
    for (int off = 1; off < 32; off <<= 1) s += __shfl_xor(s, off);
    if ((tid & 31) == 0) red[tid >> 5] = s;
    __syncthreads();
    float tot = 0.f;
#pragma unroll
    for (int i = 0; i < 8; ++i) tot += red[i];
    const float rms = rsqrtf(tot * (1.0f / DIM_) + 1e-6f);
    float4 wv = *(const float4*)&w[tid * 4];
    bf16_t* o = out + row * DIM_ + tid * 4;
    o[0] = (bf16_t)(v.x * rms * wv.x);
    o[1] = (bf16_t)(v.y * rms * wv.y);
    o[2] = (bf16_t)(v.z * rms * wv.z);
    o[3] = (bf16_t)(v.w * rms * wv.w);
}

// ---------------------------------------------------------------------------
// Double-buffered WMMA GEMM with async global->LDS pipeline.
//  C[M,N] = epilogue(A[M,K] @ Bt[N,K]^T)
//  BM=128 BN=128 BK=32, 8 waves (4 along M x 2 along N), each wave 32x64.
//  EPI 0: bf16 store               (aux unused)
//  EPI 1: f32 store, += residual   (aux = const float* residual)
//  EPI 2: bf16 store of silu(aux)*acc (aux = const bf16* gate pre-act)
// ---------------------------------------------------------------------------
template <int EPI>
__global__ __launch_bounds__(256)
void gemm_bf16_kernel(const bf16_t* __restrict__ A, const bf16_t* __restrict__ Bt,
                      void* __restrict__ Cout, const void* __restrict__ aux,
                      int M, int N, int K) {
    __shared__ bf16_t sA[2][128][40];
    __shared__ bf16_t sB[2][128][40];

    const int tid  = threadIdx.x;
    const int lane = tid & 31;
    const int wave = tid >> 5;
    const int wm = (wave >> 1) * 32;
    const int wn = (wave & 1) * 64;
    const int hl  = lane >> 4;
    const int l15 = lane & 15;

    const int m0 = blockIdx.y * 128;
    const int n0 = blockIdx.x * 128;

    const int lr = tid >> 1;               // 0..127: tile row this thread copies
    const int lc = (tid & 1) << 4;         // 0 or 16

    v8f acc[2][4] = {};

    const bf16_t* gA = &A[(size_t)(m0 + lr) * K + lc];
    const bf16_t* gB = &Bt[(size_t)(n0 + lr) * K + lc];

    auto issue_tile = [&](int buf, int k0) {
        const bf16_t* ga = gA + k0;
        const bf16_t* gb = gB + k0;
        bf16_t* la = &sA[buf][lr][lc];
        bf16_t* lb = &sB[buf][lr][lc];
        tile_b128(ga,     la);
        tile_b128(ga + 8, la + 8);
        tile_b128(gb,     lb);
        tile_b128(gb + 8, lb + 8);
    };

    const int nk = K >> 5;
    issue_tile(0, 0);

    for (int t = 0; t < nk; ++t) {
        const int buf = t & 1;
        if (t + 1 < nk) {
            issue_tile(buf ^ 1, (t + 1) << 5);   // prefetch next tile
            tile_wait_prev();                    // current tile landed (in-order)
        } else {
            tile_wait_all();
        }
        __syncthreads();

        Frag fa[2], fb[4];
#pragma unroll
        for (int i = 0; i < 2; ++i) {
            const int r = wm + i * 16 + l15;
            fa[i].u[0] = *(const u32x4*)&sA[buf][r][hl * 8];
            fa[i].u[1] = *(const u32x4*)&sA[buf][r][hl * 8 + 16];
        }
#pragma unroll
        for (int j = 0; j < 4; ++j) {
            const int r = wn + j * 16 + l15;
            fb[j].u[0] = *(const u32x4*)&sB[buf][r][hl * 8];
            fb[j].u[1] = *(const u32x4*)&sB[buf][r][hl * 8 + 16];
        }
#pragma unroll
        for (int i = 0; i < 2; ++i)
#pragma unroll
            for (int j = 0; j < 4; ++j)
                acc[i][j] = wmma_bf16(fa[i].v, fb[j].v, acc[i][j]);
        __syncthreads();     // everyone done reading buf before it is refilled
    }

    // Epilogue: C layout -> lanes 0-15: M=r, lanes 16-31: M=8+r ; N = l15
#pragma unroll
    for (int i = 0; i < 2; ++i) {
#pragma unroll
        for (int j = 0; j < 4; ++j) {
#pragma unroll
            for (int r = 0; r < 8; ++r) {
                const int m = m0 + wm + i * 16 + hl * 8 + r;
                const int n = n0 + wn + j * 16 + l15;
                const size_t idx = (size_t)m * N + n;
                const float vv = acc[i][j][r];
                if constexpr (EPI == 0) {
                    ((bf16_t*)Cout)[idx] = (bf16_t)vv;
                } else if constexpr (EPI == 1) {
                    ((float*)Cout)[idx] = ((const float*)aux)[idx] + vv;
                } else {
                    const float g = (float)((const bf16_t*)aux)[idx];
                    const float s = g / (1.f + __expf(-g));   // silu(gate)
                    ((bf16_t*)Cout)[idx] = (bf16_t)(s * vv);
                }
            }
        }
    }
}

// ---------------------------------------------------------------------------
// Flash attention: one 32-key block step (templated on causal masking)
// ---------------------------------------------------------------------------
template <bool MASK>
static __device__ __forceinline__ void flash_block(
    const int kb, const int q0, const int hl, const int l15,
    const bf16_t* __restrict__ kbase,     // &k[b*T*DIM + h*HD]
    const bf16_t* __restrict__ vtbase,    // &vt[bh*HD*T]
    const Frag (&qf)[2], bf16_t (*sPw)[40],
    float (&mrow)[8], float (&lrow)[8], v8f (&oacc)[4]) {

    // ---- S = Q @ K^T (two 16x16 tiles) ---------------------------------
    v8f s0 = {}, s1 = {};
    {
        Frag kf[2][2];
#pragma unroll
        for (int g2 = 0; g2 < 2; ++g2) {
            const bf16_t* kr = kbase + (size_t)(kb + g2 * 16 + l15) * DIM_;
#pragma unroll
            for (int s = 0; s < 2; ++s) {
                kf[g2][s].u[0] = *(const u32x4*)&kr[s * 32 + hl * 8];
                kf[g2][s].u[1] = *(const u32x4*)&kr[s * 32 + hl * 8 + 16];
            }
        }
        s0 = wmma_bf16(qf[0].v, kf[0][0].v, s0);
        s0 = wmma_bf16(qf[1].v, kf[0][1].v, s0);
        s1 = wmma_bf16(qf[0].v, kf[1][0].v, s1);
        s1 = wmma_bf16(qf[1].v, kf[1][1].v, s1);
    }

    const float sc = 0.125f;                       // 1/sqrt(64)
#pragma unroll
    for (int r = 0; r < 8; ++r) {
        float a0 = s0[r] * sc, a1 = s1[r] * sc;
        if constexpr (MASK) {
            const int qi = q0 + hl * 8 + r;
            if (kb + l15 > qi)      a0 = -1.0e30f;
            if (kb + 16 + l15 > qi) a1 = -1.0e30f;
        }
        s0[r] = a0; s1[r] = a1;
    }

    // ---- online softmax -------------------------------------------------
    float alpha[8];
#pragma unroll
    for (int r = 0; r < 8; ++r) {
        const float t = rmax16(fmaxf(s0[r], s1[r]));
        const float mnew = fmaxf(mrow[r], t);
        alpha[r] = __expf(mrow[r] - mnew);
        mrow[r] = mnew;
        const float p0 = __expf(s0[r] - mnew);
        const float p1 = __expf(s1[r] - mnew);
        s0[r] = p0; s1[r] = p1;
        lrow[r] = lrow[r] * alpha[r] + rsum16(p0 + p1);
    }
#pragma unroll
    for (int g = 0; g < 4; ++g)
#pragma unroll
        for (int r = 0; r < 8; ++r) oacc[g][r] *= alpha[r];

    // ---- P (C-layout) -> LDS -> A-layout fragment -----------------------
#pragma unroll
    for (int r = 0; r < 8; ++r) {
        const int m = hl * 8 + r;
        sPw[m][l15]      = (bf16_t)s0[r];
        sPw[m][16 + l15] = (bf16_t)s1[r];
    }
    asm volatile("" ::: "memory");   // DS ops are in-order within a wave
    Frag pf;
    pf.u[0] = *(const u32x4*)&sPw[l15][hl * 8];
    pf.u[1] = *(const u32x4*)&sPw[l15][hl * 8 + 16];

    // ---- O += P @ V (B-frags from pre-transposed Vt) --------------------
#pragma unroll
    for (int g = 0; g < 4; ++g) {
        Frag vf;
        const bf16_t* vr = vtbase + (size_t)(g * 16 + l15) * T_ + kb;
        vf.u[0] = *(const u32x4*)&vr[hl * 8];
        vf.u[1] = *(const u32x4*)&vr[hl * 8 + 16];
        oacc[g] = wmma_bf16(pf.v, vf.v, oacc[g]);
    }
}

// ---------------------------------------------------------------------------
// Per-wave causal flash attention; one wave owns a 16-row query tile.
// ---------------------------------------------------------------------------
__global__ __launch_bounds__(128)
void flash_attn_kernel(const bf16_t* __restrict__ q, const bf16_t* __restrict__ k,
                       const bf16_t* __restrict__ vt, bf16_t* __restrict__ o) {
    __shared__ bf16_t sP[4][16][40];

    const int wid  = threadIdx.x >> 5;
    const int lane = threadIdx.x & 31;
    const int hl   = lane >> 4;
    const int l15  = lane & 15;

    const int gw = blockIdx.x * 4 + wid;
    const int qtiles = T_ / 16;
    const int qt = gw % qtiles;
    const int bh = gw / qtiles;
    const int b  = bh / HEADS_;
    const int h  = bh % HEADS_;
    const int q0 = qt * 16;

    const bf16_t* kbase  = k + (size_t)(b * T_) * DIM_ + h * HD_;
    const bf16_t* vtbase = vt + (size_t)bh * HD_ * T_;

    Frag qf[2];
    {
        const bf16_t* qr = q + (size_t)(b * T_ + q0 + l15) * DIM_ + h * HD_;
#pragma unroll
        for (int s = 0; s < 2; ++s) {
            qf[s].u[0] = *(const u32x4*)&qr[s * 32 + hl * 8];
            qf[s].u[1] = *(const u32x4*)&qr[s * 32 + hl * 8 + 16];
        }
    }

    float mrow[8], lrow[8];
    v8f oacc[4] = {};
#pragma unroll
    for (int r = 0; r < 8; ++r) { mrow[r] = -3.0e38f; lrow[r] = 0.f; }

    const int nblocks = (q0 + 15 + 32) / 32;                   // causal coverage
    const int nomask  = (q0 >= 31) ? ((q0 - 31) / 32 + 1) : 0; // fully-visible blocks

    int kb = 0;
    for (; kb < nomask * 32; kb += 32)
        flash_block<false>(kb, q0, hl, l15, kbase, vtbase, qf, sP[wid], mrow, lrow, oacc);
    for (; kb < nblocks * 32; kb += 32)
        flash_block<true>(kb, q0, hl, l15, kbase, vtbase, qf, sP[wid], mrow, lrow, oacc);

    float invl[8];
#pragma unroll
    for (int r = 0; r < 8; ++r) invl[r] = 1.f / lrow[r];
#pragma unroll
    for (int g = 0; g < 4; ++g) {
#pragma unroll
        for (int r = 0; r < 8; ++r) {
            const int m = q0 + hl * 8 + r;
            o[(size_t)(b * T_ + m) * DIM_ + h * HD_ + g * 16 + l15] =
                (bf16_t)(oacc[g][r] * invl[r]);
        }
    }
}

// ---------------------------------------------------------------------------
// Orchestration
// ---------------------------------------------------------------------------
extern "C" void kernel_launch(void* const* d_in, const int* in_sizes, int n_in,
                              void* d_out, int out_size, void* d_ws, size_t ws_size,
                              hipStream_t stream) {
    (void)in_sizes; (void)n_in; (void)out_size; (void)ws_size;

    const float* x      = (const float*)d_in[0];
    const float* wq     = (const float*)d_in[1];
    const float* wk     = (const float*)d_in[2];
    const float* wv     = (const float*)d_in[3];
    const float* wo     = (const float*)d_in[4];
    const float* wgate  = (const float*)d_in[5];
    const float* wup    = (const float*)d_in[6];
    const float* wdown  = (const float*)d_in[7];
    const float* anorm  = (const float*)d_in[8];
    const float* fnorm  = (const float*)d_in[9];
    float* out = (float*)d_out;

    size_t cur = 0;
    auto alloc = [&](size_t bytes) -> void* {
        void* p = (char*)d_ws + cur;
        cur += (bytes + 255) & ~(size_t)255;
        return p;
    };
    bf16_t* wqT = (bf16_t*)alloc((size_t)DIM_ * DIM_ * 2);
    bf16_t* wkT = (bf16_t*)alloc((size_t)DIM_ * DIM_ * 2);
    bf16_t* wvT = (bf16_t*)alloc((size_t)DIM_ * DIM_ * 2);
    bf16_t* woT = (bf16_t*)alloc((size_t)DIM_ * DIM_ * 2);
    bf16_t* wgT = (bf16_t*)alloc((size_t)HID_ * DIM_ * 2);
    bf16_t* wuT = (bf16_t*)alloc((size_t)HID_ * DIM_ * 2);
    bf16_t* wdT = (bf16_t*)alloc((size_t)DIM_ * HID_ * 2);
    bf16_t* xn  = (bf16_t*)alloc((size_t)TOK_ * DIM_ * 2);
    bf16_t* qb  = (bf16_t*)alloc((size_t)TOK_ * DIM_ * 2);
    bf16_t* kbf = (bf16_t*)alloc((size_t)TOK_ * DIM_ * 2);
    bf16_t* vbf = (bf16_t*)alloc((size_t)TOK_ * DIM_ * 2);
    bf16_t* vtb = (bf16_t*)alloc((size_t)TOK_ * DIM_ * 2);
    bf16_t* ao  = (bf16_t*)alloc((size_t)TOK_ * DIM_ * 2);
    bf16_t* gbf = (bf16_t*)alloc((size_t)TOK_ * HID_ * 2);
    bf16_t* hbf = (bf16_t*)alloc((size_t)TOK_ * HID_ * 2);

    const dim3 blk256(256), blk128(128);

    transpose_cvt_kernel<<<dim3(DIM_/32, DIM_/32), blk256, 0, stream>>>(wq, wqT, DIM_, DIM_);
    transpose_cvt_kernel<<<dim3(DIM_/32, DIM_/32), blk256, 0, stream>>>(wk, wkT, DIM_, DIM_);
    transpose_cvt_kernel<<<dim3(DIM_/32, DIM_/32), blk256, 0, stream>>>(wv, wvT, DIM_, DIM_);
    transpose_cvt_kernel<<<dim3(DIM_/32, DIM_/32), blk256, 0, stream>>>(wo, woT, DIM_, DIM_);
    transpose_cvt_kernel<<<dim3(HID_/32, DIM_/32), blk256, 0, stream>>>(wgate, wgT, DIM_, HID_);
    transpose_cvt_kernel<<<dim3(HID_/32, DIM_/32), blk256, 0, stream>>>(wup,   wuT, DIM_, HID_);
    transpose_cvt_kernel<<<dim3(DIM_/32, HID_/32), blk256, 0, stream>>>(wdown, wdT, HID_, DIM_);

    rmsnorm_kernel<<<dim3(TOK_), blk256, 0, stream>>>(x, anorm, xn);

    gemm_bf16_kernel<0><<<dim3(DIM_/128, TOK_/128), blk256, 0, stream>>>(xn, wqT, qb,  nullptr, TOK_, DIM_, DIM_);
    gemm_bf16_kernel<0><<<dim3(DIM_/128, TOK_/128), blk256, 0, stream>>>(xn, wkT, kbf, nullptr, TOK_, DIM_, DIM_);
    gemm_bf16_kernel<0><<<dim3(DIM_/128, TOK_/128), blk256, 0, stream>>>(xn, wvT, vbf, nullptr, TOK_, DIM_, DIM_);

    transpose_v_kernel<<<dim3(T_/32, HD_/32, B_*HEADS_), blk256, 0, stream>>>(vbf, vtb);

    flash_attn_kernel<<<dim3(B_*HEADS_*(T_/16)/4), blk128, 0, stream>>>(qb, kbf, vtb, ao);

    gemm_bf16_kernel<1><<<dim3(DIM_/128, TOK_/128), blk256, 0, stream>>>(ao, woT, out, x, TOK_, DIM_, DIM_);

    rmsnorm_kernel<<<dim3(TOK_), blk256, 0, stream>>>(out, fnorm, xn);

    gemm_bf16_kernel<0><<<dim3(HID_/128, TOK_/128), blk256, 0, stream>>>(xn, wgT, gbf, nullptr, TOK_, HID_, DIM_);
    gemm_bf16_kernel<2><<<dim3(HID_/128, TOK_/128), blk256, 0, stream>>>(xn, wuT, hbf, gbf, TOK_, HID_, DIM_);
    gemm_bf16_kernel<1><<<dim3(DIM_/128, TOK_/128), blk256, 0, stream>>>(hbf, wdT, out, out, TOK_, DIM_, HID_);
}
